// PointNetPlusPlusBackbone_9698036154801
// MI455X (gfx1250) — compile-verified, matrix operand order checked by reading, not run on hardware
//
#include <hip/hip_runtime.h>
#include <hip/hip_bf16.h>
#include <cstdint>
#include <cstddef>

// ---------------------------------------------------------------------------
// PointNet++ backbone for MI455X (gfx1250, wave32).
// All MLP layers run as f16 WMMA GEMMs (v_wmma_f32_16x16x32_f16) with fused
// BN (eps=1e-5) + ReLU epilogue. Each wave computes a 16x64 output strip
// (1 A fragment x 4 B fragments -> 4 accumulators) to quarter the
// activation-read traffic on the memory-bound layers.
// ---------------------------------------------------------------------------

typedef __attribute__((ext_vector_type(16))) _Float16 v16h;
typedef __attribute__((ext_vector_type(8)))  _Float16 v8h;
typedef __attribute__((ext_vector_type(8)))  float    v8f;

static inline int div_up_i(int a, int b) { return (a + b - 1) / b; }

#define PNB 8
#define PNN 16384

// -------------------------- input build ------------------------------------
__global__ void k_build_inputs(const float* __restrict__ xyz,   // [B,3,N]
                               const float* __restrict__ fea,   // [B,5,N]
                               float* __restrict__ xyzT,        // [B,N,3]
                               float* __restrict__ pts0,        // [B,N,8]
                               int B, int N) {
  int i = blockIdx.x * blockDim.x + threadIdx.x;
  if (i >= B * N) return;
  int b = i / N, n = i % N;
  float x = xyz[((size_t)b * 3 + 0) * N + n];
  float y = xyz[((size_t)b * 3 + 1) * N + n];
  float z = xyz[((size_t)b * 3 + 2) * N + n];
  xyzT[(size_t)i * 3 + 0] = x;
  xyzT[(size_t)i * 3 + 1] = y;
  xyzT[(size_t)i * 3 + 2] = z;
  pts0[(size_t)i * 8 + 0] = x;
  pts0[(size_t)i * 8 + 1] = y;
  pts0[(size_t)i * 8 + 2] = z;
#pragma unroll
  for (int c = 0; c < 5; ++c)
    pts0[(size_t)i * 8 + 3 + c] = fea[((size_t)b * 5 + c) * N + n];
}

// ------------------------ weight conversion (f32 -> f16, K-pad) ------------
__global__ void k_conv_w(const float* __restrict__ W, _Float16* __restrict__ Wh,
                         int O, int K, int Kpad) {
  int i = blockIdx.x * blockDim.x + threadIdx.x;
  if (i >= O * Kpad) return;
  int o = i / Kpad, k = i % Kpad;
  Wh[i] = (k < K) ? (_Float16)W[(size_t)o * K + k] : (_Float16)0.0f;
}

// -------------------------- farthest point sampling ------------------------
__global__ void k_fps(const float* __restrict__ xyzT, int* __restrict__ out_idx,
                      float* __restrict__ dist, int N, int npoint) {
  int b = blockIdx.x;
  int tid = threadIdx.x;
  __shared__ float s_val[256];
  __shared__ int   s_id[256];
  __shared__ int   s_far;
  const float* P = xyzT + (size_t)b * N * 3;
  float* D = dist + (size_t)b * N;
  for (int i = tid; i < N; i += 256) D[i] = 1e10f;
  if (tid == 0) s_far = 0;
  __syncthreads();
  for (int it = 0; it < npoint; ++it) {
    int far = s_far;
    if (tid == 0) out_idx[b * npoint + it] = far;
    float cx = P[(size_t)far * 3 + 0];
    float cy = P[(size_t)far * 3 + 1];
    float cz = P[(size_t)far * 3 + 2];
    float lbest = -1.0f; int lidx = 0;
    for (int i = tid; i < N; i += 256) {
      float dx = P[(size_t)i * 3 + 0] - cx;
      float dy = P[(size_t)i * 3 + 1] - cy;
      float dz = P[(size_t)i * 3 + 2] - cz;
      float d2 = dx * dx + dy * dy + dz * dz;
      float nd = fminf(D[i], d2);
      D[i] = nd;
      if (nd > lbest) { lbest = nd; lidx = i; }
    }
    s_val[tid] = lbest; s_id[tid] = lidx;
    __syncthreads();
    for (int s = 128; s > 0; s >>= 1) {
      if (tid < s && s_val[tid + s] > s_val[tid]) {
        s_val[tid] = s_val[tid + s]; s_id[tid] = s_id[tid + s];
      }
      __syncthreads();
    }
    if (tid == 0) s_far = s_id[0];
    __syncthreads();
  }
}

__global__ void k_gather_xyz(const float* __restrict__ xyzT, const int* __restrict__ idx,
                             float* __restrict__ nx, int B, int np, int N) {
  int i = blockIdx.x * blockDim.x + threadIdx.x;
  if (i >= B * np) return;
  int b = i / np;
  int id = idx[i];
#pragma unroll
  for (int c = 0; c < 3; ++c)
    nx[(size_t)i * 3 + c] = xyzT[((size_t)b * N + id) * 3 + c];
}

// ------------------------------ ball query ---------------------------------
__global__ void k_ball(const float* __restrict__ xyzT, const float* __restrict__ nx,
                       int* __restrict__ idx, int B, int N, int np, int ns, float thr) {
  int q = blockIdx.x * blockDim.x + threadIdx.x;
  if (q >= B * np) return;
  int b = q / np;
  const float* P = xyzT + (size_t)b * N * 3;
  float cx = nx[(size_t)q * 3 + 0];
  float cy = nx[(size_t)q * 3 + 1];
  float cz = nx[(size_t)q * 3 + 2];
  int* out = idx + (size_t)q * ns;
  int cnt = 0, first = -1;
  for (int i = 0; i < N && cnt < ns; ++i) {
    float dx = P[(size_t)i * 3 + 0] - cx;
    float dy = P[(size_t)i * 3 + 1] - cy;
    float dz = P[(size_t)i * 3 + 2] - cz;
    float d2 = dx * dx + dy * dy + dz * dz;
    if (d2 < thr) { if (first < 0) first = i; out[cnt++] = i; }
  }
  if (first < 0) first = 0;
  for (int k = cnt; k < ns; ++k) out[k] = first;
}

// ------------------------------ grouping -----------------------------------
__global__ void k_group(const float* __restrict__ xyzT, const float* __restrict__ nx,
                        const int* __restrict__ idx, const float* __restrict__ PTS,
                        int Cp, _Float16* __restrict__ act,
                        int B, int N, int np, int ns, int Kpad) {
  int i = blockIdx.x * blockDim.x + threadIdx.x;
  int total = B * np * ns * Kpad;
  if (i >= total) return;
  int c = i % Kpad;
  int r = i / Kpad;            // global sample row = (b*np + j)*ns + s
  int s = r % ns;
  int q = r / ns;              // b*np + j
  int b = q / np;
  int pid = idx[(size_t)q * ns + s];
  float v = 0.0f;
  if (c < 3)
    v = xyzT[((size_t)b * N + pid) * 3 + c] - nx[(size_t)q * 3 + c];
  else if (c < 3 + Cp)
    v = PTS[((size_t)b * N + pid) * Cp + (c - 3)];
  act[(size_t)i] = (_Float16)v;
}

__global__ void k_group_all(const float* __restrict__ xyz,   // [B,S,3]
                            const float* __restrict__ pts,   // [B,S,Cp]
                            _Float16* __restrict__ act,
                            int B, int S, int Cp, int Kpad) {
  int i = blockIdx.x * blockDim.x + threadIdx.x;
  int total = B * S * Kpad;
  if (i >= total) return;
  int c = i % Kpad;
  int r = i / Kpad;            // b*S + s
  float v = 0.0f;
  if (c < 3)            v = xyz[(size_t)r * 3 + c];
  else if (c < 3 + Cp)  v = pts[(size_t)r * Cp + (c - 3)];
  act[(size_t)i] = (_Float16)v;
}

// ---------------- WMMA GEMM + BatchNorm + ReLU (the compute core) ----------
// D = relu( (A @ W^T + bias) * gamma/sqrt(1+eps) + beta )
// A: [M, Kpad] f16 row-major, W: [Nout, Kpad] f16 row-major, Nout % 64 == 0.
// One wave computes a 16x64 output strip: one A fragment is reused against
// four B fragments per K-step (4x v_wmma_f32_16x16x32_f16).
__global__ void __launch_bounds__(256)
k_wmma_gemm_bn_relu(const _Float16* __restrict__ A, const _Float16* __restrict__ W,
                    const float* __restrict__ bias, const float* __restrict__ gamma,
                    const float* __restrict__ beta,
                    _Float16* __restrict__ O16, float* __restrict__ O32,
                    int M, int Kpad, int Nout) {
  const int lane = threadIdx.x & 31;
  const int strip = blockIdx.x * 8 + (threadIdx.x >> 5);
  const int stripsN = Nout >> 6;          // strips of 4 x 16 output columns
  const int tilesM = M >> 4;
  if (strip >= tilesM * stripsN) return;
  const int tm  = strip / stripsN;
  const int tn4 = strip % stripsN;
  const int half = lane >> 4;      // K sub-range select (ISA 16-bit A/B layout)
  const int lid  = lane & 15;      // M index (A) / N index (B)
  const _Float16* arow = A + (size_t)(tm * 16 + lid) * Kpad + half * 8;
  const _Float16* wr[4];
#pragma unroll
  for (int j = 0; j < 4; ++j)
    wr[j] = W + (size_t)(tn4 * 64 + j * 16 + lid) * Kpad + half * 8;

  v8f acc[4] = {v8f{}, v8f{}, v8f{}, v8f{}};
  for (int k = 0; k < Kpad; k += 32) {
    if (k + 128 < Kpad) {
      __builtin_prefetch(arow + k + 128, 0, 1);
      __builtin_prefetch(wr[0] + k + 128, 0, 1);
    }
    v8h a0 = *(const v8h*)(arow + k);        // K = k + half*8 + [0..7]
    v8h a1 = *(const v8h*)(arow + k + 16);   // K = k + 16 + half*8 + [0..7]
    v16h av;
#pragma unroll
    for (int e = 0; e < 8; ++e) { av[e] = a0[e]; av[e + 8] = a1[e]; }
#pragma unroll
    for (int j = 0; j < 4; ++j) {
      v8h b0 = *(const v8h*)(wr[j] + k);
      v8h b1 = *(const v8h*)(wr[j] + k + 16);
      v16h bv;
#pragma unroll
      for (int e = 0; e < 8; ++e) { bv[e] = b0[e]; bv[e + 8] = b1[e]; }
      acc[j] = __builtin_amdgcn_wmma_f32_16x16x32_f16(
          /*neg_a=*/false, av, /*neg_b=*/false, bv,
          /*c_mod=*/(short)0, acc[j], /*reuse_a=*/false, /*reuse_b=*/false);
    }
  }
  const float inv = rsqrtf(1.0f + 1e-5f);
#pragma unroll
  for (int j = 0; j < 4; ++j) {
    const int n = tn4 * 64 + j * 16 + lid;
    const float sc = gamma[n] * inv;
    const float bs = bias[n];
    const float bt = beta[n];
#pragma unroll
    for (int r = 0; r < 8; ++r) {
      int row = tm * 16 + r + half * 8;      // ISA C/D layout
      float v = (acc[j][r] + bs) * sc + bt;
      v = v > 0.0f ? v : 0.0f;
      if (O16) O16[(size_t)row * Nout + n] = (_Float16)v;
      if (O32) O32[(size_t)row * Nout + n] = v;
    }
  }
}

// ------------------------------ max pooling --------------------------------
__global__ void k_pool(const _Float16* __restrict__ F, float* __restrict__ pooled,
                       float* __restrict__ outT, int B, int np, int ns, int C) {
  int i = blockIdx.x * blockDim.x + threadIdx.x;
  if (i >= B * np * C) return;
  int c = i % C;
  int q = i / C;                 // b*np + j
  int b = q / np, j = q % np;
  const _Float16* base = F + (size_t)q * ns * C + c;
  float m = -3.4e38f;
  for (int s = 0; s < ns; ++s) m = fmaxf(m, (float)base[(size_t)s * C]);
  pooled[(size_t)i] = m;
  if (outT) outT[(size_t)b * C * np + (size_t)c * np + j] = m;
}

// ------------------------------ 3-NN ---------------------------------------
__global__ void k_three_nn(const float* __restrict__ xyz1, const float* __restrict__ xyz2,
                           int* __restrict__ idx3, float* __restrict__ w3,
                           int B, int N1, int S) {
  int i = blockIdx.x * blockDim.x + threadIdx.x;
  if (i >= B * N1) return;
  int b = i / N1;
  float x = xyz1[(size_t)i * 3 + 0];
  float y = xyz1[(size_t)i * 3 + 1];
  float z = xyz1[(size_t)i * 3 + 2];
  const float* Q = xyz2 + (size_t)b * S * 3;
  float d0 = 1e30f, d1 = 1e30f, d2v = 1e30f;
  int i0 = 0, i1 = 0, i2 = 0;
  for (int s = 0; s < S; ++s) {
    float dx = Q[(size_t)s * 3 + 0] - x;
    float dy = Q[(size_t)s * 3 + 1] - y;
    float dz = Q[(size_t)s * 3 + 2] - z;
    float d = dx * dx + dy * dy + dz * dz;
    if (d < d0)      { d2v = d1; i2 = i1; d1 = d0; i1 = i0; d0 = d; i0 = s; }
    else if (d < d1) { d2v = d1; i2 = i1; d1 = d;  i1 = s; }
    else if (d < d2v){ d2v = d;  i2 = s; }
  }
  float r0 = 1.0f / (sqrtf(fmaxf(d0, 0.0f)) + 1e-8f);
  float r1 = 1.0f / (sqrtf(fmaxf(d1, 0.0f)) + 1e-8f);
  float r2 = 1.0f / (sqrtf(fmaxf(d2v, 0.0f)) + 1e-8f);
  float sum = r0 + r1 + r2;
  idx3[(size_t)i * 3 + 0] = i0; w3[(size_t)i * 3 + 0] = r0 / sum;
  idx3[(size_t)i * 3 + 1] = i1; w3[(size_t)i * 3 + 1] = r1 / sum;
  idx3[(size_t)i * 3 + 2] = i2; w3[(size_t)i * 3 + 2] = r2 / sum;
}

// --------------- FP-stage activation build (concat + interp) ---------------
__global__ void k_fp_act(const float* __restrict__ P1, int p1_stride, int p1_off, int C1,
                         const float* __restrict__ P2, int S2, int C2,
                         const int* __restrict__ idx3, const float* __restrict__ w3,
                         int bcast, _Float16* __restrict__ act,
                         int B, int N1, int Kpad) {
  int i = blockIdx.x * blockDim.x + threadIdx.x;
  int total = B * N1 * Kpad;
  if (i >= total) return;
  int c = i % Kpad;
  int r = i / Kpad;              // b*N1 + n
  int b = r / N1;
  float v = 0.0f;
  if (c < C1) {
    v = P1[(size_t)r * p1_stride + p1_off + c];
  } else if (c < C1 + C2) {
    int cc = c - C1;
    const float* Q = P2 + (size_t)b * S2 * C2;
    if (bcast) {
      v = Q[cc];
    } else {
      int j0 = idx3[(size_t)r * 3 + 0];
      int j1 = idx3[(size_t)r * 3 + 1];
      int j2 = idx3[(size_t)r * 3 + 2];
      v = w3[(size_t)r * 3 + 0] * Q[(size_t)j0 * C2 + cc]
        + w3[(size_t)r * 3 + 1] * Q[(size_t)j1 * C2 + cc]
        + w3[(size_t)r * 3 + 2] * Q[(size_t)j2 * C2 + cc];
    }
  }
  act[(size_t)i] = (_Float16)v;
}

// -------------- final per-point features: transpose + global max -----------
__global__ void k_pf_gf(const _Float16* __restrict__ F, float* __restrict__ pf,
                        float* __restrict__ gf, int N) {
  int b = blockIdx.x >> 7;
  int ch = blockIdx.x & 127;
  int tid = threadIdx.x;
  __shared__ float s[256];
  float m = -3.4e38f;
  for (int n = tid; n < N; n += 256) {
    float v = (float)F[((size_t)b * N + n) * 128 + ch];
    pf[(size_t)b * 128 * N + (size_t)ch * N + n] = v;
    m = fmaxf(m, v);
  }
  s[tid] = m;
  __syncthreads();
  for (int st = 128; st > 0; st >>= 1) {
    if (tid < st) s[tid] = fmaxf(s[tid], s[tid + st]);
    __syncthreads();
  }
  if (tid == 0) gf[b * 128 + ch] = s[0];
}

// ------------------------- small transpose / fill --------------------------
__global__ void k_transpose(const float* __restrict__ in, float* __restrict__ out,
                            int B, int Np, int C) {
  int i = blockIdx.x * blockDim.x + threadIdx.x;
  if (i >= B * Np * C) return;
  int c = i % C;
  int q = i / C;
  int b = q / Np, j = q % Np;
  out[(size_t)b * C * Np + (size_t)c * Np + j] = in[(size_t)i];
}

__global__ void k_zero(float* __restrict__ p, int n) {
  int i = blockIdx.x * blockDim.x + threadIdx.x;
  if (i < n) p[i] = 0.0f;
}

// ---------------------------------------------------------------------------
extern "C" void kernel_launch(void* const* d_in, const int* in_sizes, int n_in,
                              void* d_out, int out_size, void* d_ws, size_t ws_size,
                              hipStream_t stream) {
  (void)in_sizes; (void)n_in; (void)out_size; (void)ws_size;
  const int B = PNB, N = PNN;
  const float* xyz = (const float*)d_in[0];
  const float* fea = (const float*)d_in[1];
  float* out = (float*)d_out;

  // ---- output layout (floats), concatenated in reference return order ----
  const size_t OFF_PF  = 0;
  const size_t OFF_GF  = OFF_PF  + (size_t)B * 128 * N;
  const size_t OFF_L1X = OFF_GF  + (size_t)B * 128;
  const size_t OFF_L1P = OFF_L1X + (size_t)B * 3 * 512;
  const size_t OFF_L2X = OFF_L1P + (size_t)B * 128 * 512;
  const size_t OFF_L2P = OFF_L2X + (size_t)B * 3 * 128;
  const size_t OFF_L3X = OFF_L2P + (size_t)B * 256 * 128;
  const size_t OFF_L3P = OFF_L3X + (size_t)B * 3;

  // ---- workspace carve-out ----
  char* ws = (char*)d_ws;
  size_t off = 0;
  auto carve = [&](size_t bytes) -> char* {
    char* p = ws + off;
    off += (bytes + 255) & ~(size_t)255;
    return p;
  };
  _Float16* bufA = (_Float16*)carve((size_t)131072 * 160 * 2);  // ping
  _Float16* bufB = (_Float16*)carve((size_t)131072 * 160 * 2);  // pong
  float* xyzT = (float*)carve((size_t)B * N * 3 * 4);
  float* pts0 = (float*)carve((size_t)B * N * 8 * 4);
  float* dist = (float*)carve((size_t)B * N * 4);
  int*   fps1 = (int*)carve((size_t)B * 512 * 4);
  float* l1x  = (float*)carve((size_t)B * 512 * 3 * 4);
  int*   bq1  = (int*)carve((size_t)B * 512 * 32 * 4);
  float* l1p  = (float*)carve((size_t)B * 512 * 128 * 4);
  int*   fps2 = (int*)carve((size_t)B * 128 * 4);
  float* l2x  = (float*)carve((size_t)B * 128 * 3 * 4);
  int*   bq2  = (int*)carve((size_t)B * 128 * 64 * 4);
  float* l2p  = (float*)carve((size_t)B * 128 * 256 * 4);
  float* l3p  = (float*)carve((size_t)B * 1024 * 4);
  float* u2f  = (float*)carve((size_t)B * 128 * 256 * 4);
  float* u1f  = (float*)carve((size_t)B * 512 * 128 * 4);
  int*   idx3 = (int*)carve((size_t)B * N * 3 * 4);
  float* w3   = (float*)carve((size_t)B * N * 3 * 4);
  _Float16* wh = (_Float16*)carve((size_t)2000000 * 2);

  // ---- layer table: d_in index of W, (O, K, Kpad). Order follows params. ----
  struct LayerSpec { int wi; int O; int K; int Kpad; };
  const LayerSpec L[17] = {
    { 2,   64,   11,   32}, { 6,   64,   64,   64}, {10,  128,   64,   64}, // sa1
    {14,  128,  131,  160}, {18,  128,  128,  128}, {22,  256,  128,  128}, // sa2
    {26,  256,  259,  288}, {30,  512,  256,  256}, {34, 1024,  512,  512}, // sa3
    {38,  256, 1280, 1280}, {42,  256,  256,  256},                          // fp3
    {46,  256,  384,  384}, {50,  128,  256,  256},                          // fp2
    {54,  128,  133,  160}, {58,  128,  128,  128}, {62,  128,  128,  128}, // fp1
    {66,  128,  128,  128}                                                   // final
  };
  _Float16* WH[17];
  {
    size_t woff = 0;
    for (int i = 0; i < 17; ++i) {
      WH[i] = wh + woff;
      woff += (size_t)L[i].O * L[i].Kpad;
      int n = L[i].O * L[i].Kpad;
      k_conv_w<<<div_up_i(n, 256), 256, 0, stream>>>(
          (const float*)d_in[L[i].wi], WH[i], L[i].O, L[i].K, L[i].Kpad);
    }
  }

  auto gemm = [&](const _Float16* Ap, int li, _Float16* O16, float* O32, int M) {
    const LayerSpec& s = L[li];
    int strips = (M / 16) * (s.O / 64);     // one wave per 16x64 output strip
    k_wmma_gemm_bn_relu<<<div_up_i(strips, 8), 256, 0, stream>>>(
        Ap, WH[li],
        (const float*)d_in[s.wi + 1],   // bias
        (const float*)d_in[s.wi + 2],   // gamma
        (const float*)d_in[s.wi + 3],   // beta
        O16, O32, M, s.Kpad, s.O);
  };

  // ---- build pts0 / xyzT ----
  k_build_inputs<<<div_up_i(B * N, 256), 256, 0, stream>>>(xyz, fea, xyzT, pts0, B, N);

  // ================= SA1: npoint=512, ns=32, 11 -> 64 -> 64 -> 128 ========
  k_fps<<<B, 256, 0, stream>>>(xyzT, fps1, dist, N, 512);
  k_gather_xyz<<<div_up_i(B * 512, 256), 256, 0, stream>>>(xyzT, fps1, l1x, B, 512, N);
  k_ball<<<div_up_i(B * 512, 256), 256, 0, stream>>>(xyzT, l1x, bq1, B, N, 512, 32, 0.0016f);
  k_group<<<div_up_i(B * 512 * 32 * 32, 256), 256, 0, stream>>>(
      xyzT, l1x, bq1, pts0, 8, bufA, B, N, 512, 32, 32);
  const int M1 = B * 512 * 32;
  gemm(bufA, 0, bufB, nullptr, M1);
  gemm(bufB, 1, bufA, nullptr, M1);
  gemm(bufA, 2, bufB, nullptr, M1);
  k_pool<<<div_up_i(B * 512 * 128, 256), 256, 0, stream>>>(
      bufB, l1p, out + OFF_L1P, B, 512, 32, 128);

  // ================= SA2: npoint=128, ns=64, 131 -> 128 -> 128 -> 256 =====
  k_fps<<<B, 256, 0, stream>>>(l1x, fps2, dist, 512, 128);
  k_gather_xyz<<<div_up_i(B * 128, 256), 256, 0, stream>>>(l1x, fps2, l2x, B, 128, 512);
  k_ball<<<div_up_i(B * 128, 256), 256, 0, stream>>>(l1x, l2x, bq2, B, 512, 128, 64, 0.0256f);
  k_group<<<div_up_i(B * 128 * 64 * 160, 256), 256, 0, stream>>>(
      l1x, l2x, bq2, l1p, 128, bufA, B, 512, 128, 64, 160);
  const int M2 = B * 128 * 64;
  gemm(bufA, 3, bufB, nullptr, M2);
  gemm(bufB, 4, bufA, nullptr, M2);
  gemm(bufA, 5, bufB, nullptr, M2);
  k_pool<<<div_up_i(B * 128 * 256, 256), 256, 0, stream>>>(
      bufB, l2p, out + OFF_L2P, B, 128, 64, 256);

  // ================= SA3 (group_all): 259 -> 256 -> 512 -> 1024 ===========
  k_group_all<<<div_up_i(B * 128 * 288, 256), 256, 0, stream>>>(
      l2x, l2p, bufA, B, 128, 256, 288);
  const int M3 = B * 128;
  gemm(bufA, 6, bufB, nullptr, M3);
  gemm(bufB, 7, bufA, nullptr, M3);
  gemm(bufA, 8, bufB, nullptr, M3);
  k_pool<<<div_up_i(B * 1024, 256), 256, 0, stream>>>(
      bufB, l3p, out + OFF_L3P, B, 1, 128, 1024);

  // ================= FP3: broadcast l3p, concat l2p, 1280 -> 256 -> 256 ===
  k_fp_act<<<div_up_i(B * 128 * 1280, 256), 256, 0, stream>>>(
      l2p, 256, 0, 256, l3p, 1, 1024, nullptr, nullptr, 1, bufA, B, 128, 1280);
  gemm(bufA, 9, bufB, nullptr, M3);
  gemm(bufB, 10, bufA, u2f, M3);

  // ================= FP2: 3-NN(l1x<-l2x), concat l1p, 384 -> 256 -> 128 ===
  k_three_nn<<<div_up_i(B * 512, 256), 256, 0, stream>>>(l1x, l2x, idx3, w3, B, 512, 128);
  k_fp_act<<<div_up_i(B * 512 * 384, 256), 256, 0, stream>>>(
      l1p, 128, 0, 128, u2f, 128, 256, idx3, w3, 0, bufA, B, 512, 384);
  const int M4 = B * 512;
  gemm(bufA, 11, bufB, nullptr, M4);
  gemm(bufB, 12, bufA, u1f, M4);

  // ================= FP1: 3-NN(xyzT<-l1x), concat feat(5), 133(160)->128x3 =
  k_three_nn<<<div_up_i(B * N, 256), 256, 0, stream>>>(xyzT, l1x, idx3, w3, B, N, 512);
  k_fp_act<<<div_up_i(B * N * 160, 256), 256, 0, stream>>>(
      pts0, 8, 3, 5, u1f, 512, 128, idx3, w3, 0, bufA, B, N, 160);
  const int M5 = B * N;
  gemm(bufA, 13, bufB, nullptr, M5);
  gemm(bufB, 14, bufA, nullptr, M5);
  gemm(bufA, 15, bufB, nullptr, M5);

  // ================= final layer + pf/gf ==================================
  gemm(bufB, 16, bufA, nullptr, M5);
  k_pf_gf<<<B * 128, 256, 0, stream>>>(bufA, out + OFF_PF, out + OFF_GF, N);

  // ================= small coordinate outputs =============================
  k_transpose<<<div_up_i(B * 512 * 3, 256), 256, 0, stream>>>(l1x, out + OFF_L1X, B, 512, 3);
  k_transpose<<<div_up_i(B * 128 * 3, 256), 256, 0, stream>>>(l2x, out + OFF_L2X, B, 128, 3);
  k_zero<<<1, 32, 0, stream>>>(out + OFF_L3X, B * 3);
}
